// BidirectionalMamba_73151882985990
// MI455X (gfx1250) — matrix operand order repaired and checked by reading, not compiled, not run on hardware
//
#include <hip/hip_runtime.h>
#include <hip/hip_bf16.h>

#define B_   2
#define L_   4096
#define DM_  768
#define DI_  1536
#define DS_  16
#define DTR_ 48   /* padded to 64 for WMMA K */

typedef __attribute__((ext_vector_type(16))) __bf16 v16bf;
typedef __attribute__((ext_vector_type(8)))  __bf16 v8bf;
typedef __attribute__((ext_vector_type(8)))  float  v8f;

union FragBF { v16bf v; v8bf h[2]; };

__device__ __forceinline__ float siluf(float x) { return x / (1.f + __expf(-x)); }

// ---------------------------------------------------------------------------
// Generic f32 -> bf16 convert with zero padding: out[Rout,Kout] from in[Rin,Kin]
// ---------------------------------------------------------------------------
__global__ void pad2d_f32_bf16(const float* __restrict__ in, __bf16* __restrict__ out,
                               int Rin, int Kin, int Rout, int Kout) {
    long idx = (long)blockIdx.x * blockDim.x + threadIdx.x;
    long tot = (long)Rout * Kout;
    if (idx >= tot) return;
    int  k = (int)(idx % Kout);
    long r = idx / Kout;
    float v = (r < (long)Rin && k < Kin) ? in[r * Kin + k] : 0.f;
    out[idx] = (__bf16)v;
}

// ---------------------------------------------------------------------------
// NT GEMM, bf16 inputs, f32 accumulate via v_wmma_f32_16x16x32_bf16.
// C[b, m, n] = sum_k A[b*sAb + m*K + k] * B[b*sBb + n*K + k]
// Each wave computes a 32x32 (MxN) tile with 4 accumulators.
// Fragment loads follow the CDNA5 ISA VGPR layouts (CDNA5 ISA 7.12.2).
// ---------------------------------------------------------------------------
__global__ void gemm_nt_bf16(const __bf16* __restrict__ A, const __bf16* __restrict__ Bm,
                             float* __restrict__ C,
                             int M, int N, int K, long sAb, long sBb,
                             int tilesM, int tilesN, long totalTiles) {
    int  lane = threadIdx.x & 31;
    int  wave = threadIdx.x >> 5;
    long tile = (long)blockIdx.x * (blockDim.x >> 5) + wave;
    if (tile >= totalTiles) return;
    long tpb = (long)tilesM * tilesN;
    int  b   = (int)(tile / tpb);
    long rem = tile - (long)b * tpb;
    int  tm  = (int)(rem / tilesN);
    int  tn  = (int)(rem - (long)tm * tilesN);
    int  m0 = tm * 32, n0 = tn * 32;
    int  ml = lane & 15, hi = lane >> 4;

    // A frag: lane (ml) holds row m; hi selects K chunks {0..7,16..23} vs {8..15,24..31}
    const __bf16* a0 = A + (long)b * sAb + (long)(m0 + ml) * K;
    const __bf16* a1 = a0 + (long)16 * K;
    // B frag: lane (ml) holds column n; hi selects K chunks {0..7,16..23} vs {8..15,24..31}
    const __bf16* b0 = Bm + (long)b * sBb + (long)(n0 + ml) * K;
    const __bf16* b1 = b0 + (long)16 * K;

    v8f acc00 = {}, acc01 = {}, acc10 = {}, acc11 = {};
    for (int k0 = 0; k0 < K; k0 += 32) {
        FragBF fa0, fa1, fb0, fb1;
        fa0.h[0] = *(const v8bf*)(a0 + k0 + hi * 8);
        fa0.h[1] = *(const v8bf*)(a0 + k0 + 16 + hi * 8);
        fa1.h[0] = *(const v8bf*)(a1 + k0 + hi * 8);
        fa1.h[1] = *(const v8bf*)(a1 + k0 + 16 + hi * 8);
        fb0.h[0] = *(const v8bf*)(b0 + k0 + hi * 8);
        fb0.h[1] = *(const v8bf*)(b0 + k0 + 16 + hi * 8);
        fb1.h[0] = *(const v8bf*)(b1 + k0 + hi * 8);
        fb1.h[1] = *(const v8bf*)(b1 + k0 + 16 + hi * 8);
        acc00 = __builtin_amdgcn_wmma_f32_16x16x32_bf16(false, fa0.v, false, fb0.v, (short)0, acc00, false, false);
        acc01 = __builtin_amdgcn_wmma_f32_16x16x32_bf16(false, fa0.v, false, fb1.v, (short)0, acc01, false, false);
        acc10 = __builtin_amdgcn_wmma_f32_16x16x32_bf16(false, fa1.v, false, fb0.v, (short)0, acc10, false, false);
        acc11 = __builtin_amdgcn_wmma_f32_16x16x32_bf16(false, fa1.v, false, fb1.v, (short)0, acc11, false, false);
    }

    // C/D layout: VGPR r -> M = hi*8 + r, N = lane&15
    float* c = C + (long)b * M * N;
    int n = n0 + ml;
#pragma unroll
    for (int r2 = 0; r2 < 8; ++r2) {
        int mr = hi * 8 + r2;
        c[(long)(m0 + mr) * N + n]           = acc00[r2];
        c[(long)(m0 + mr) * N + n + 16]      = acc01[r2];
        c[(long)(m0 + 16 + mr) * N + n]      = acc10[r2];
        c[(long)(m0 + 16 + mr) * N + n + 16] = acc11[r2];
    }
}

// ---------------------------------------------------------------------------
// Depthwise causal conv (k=4) + SiLU on the x half of xz; writes transposed
// bf16 [B, L, DI] (K-contiguous for the following WMMA GEMM; also the scan's u).
// reverse=1 operates on the L-reversed sequence (branch b).
// ---------------------------------------------------------------------------
__global__ void conv_silu_kernel(const float* __restrict__ xz, const float* __restrict__ w,
                                 const float* __restrict__ bias, __bf16* __restrict__ xcT,
                                 int reverse) {
    long idx = (long)blockIdx.x * blockDim.x + threadIdx.x;
    long tot = (long)B_ * DI_ * L_;
    if (idx >= tot) return;
    int  l = (int)(idx % L_);
    long t = idx / L_;
    int  d = (int)(t % DI_);
    int  b = (int)(t / DI_);
    const float* row = xz + ((long)b * 2 * DI_ + d) * L_;
    float acc = bias[d];
#pragma unroll
    for (int j = 0; j < 4; ++j) {
        int li = l - 3 + j;
        if (li >= 0) {
            int src = reverse ? (L_ - 1 - li) : li;
            acc += w[d * 4 + j] * row[src];
        }
    }
    acc = siluf(acc);
    xcT[((long)b * L_ + l) * DI_ + d] = (__bf16)acc;
}

// ---------------------------------------------------------------------------
// Split dbl [B,96,L] into dt_low (padded bf16 [B,L,64]) and B/C ([B,L,32] f32).
// ---------------------------------------------------------------------------
__global__ void pack_dbl_kernel(const float* __restrict__ dbl, __bf16* __restrict__ dtlT,
                                float* __restrict__ bcT) {
    long idx = (long)blockIdx.x * blockDim.x + threadIdx.x;
    long tot = (long)B_ * L_ * 64;
    if (idx >= tot) return;
    int e = (int)(idx % 64);
    int l = (int)((idx / 64) % L_);
    int b = (int)(idx / ((long)64 * L_));
    float v = (e < DTR_) ? dbl[((long)b * 96 + e) * L_ + l] : 0.f;
    dtlT[idx] = (__bf16)v;
    if (e < 32)
        bcT[((long)b * L_ + l) * 32 + e] = dbl[((long)b * 96 + DTR_ + e) * L_ + l];
}

// ---------------------------------------------------------------------------
// Selective scan. One thread per (b, d, state n): 16 lanes share a channel;
// per-step reduction over states via shfl_xor. Branch b (reverse=1) reads z at
// the reversed index and accumulates into y at the reversed index.
// ---------------------------------------------------------------------------
__global__ void scan_kernel(const float* __restrict__ dt, const __bf16* __restrict__ xcT,
                            const float* __restrict__ bcT, const float* __restrict__ xz,
                            const float* __restrict__ A_log, const float* __restrict__ dt_bias,
                            const float* __restrict__ Dv, float* __restrict__ y, int reverse) {
    int tid = blockIdx.x * blockDim.x + threadIdx.x;
    int n  = tid & 15;
    int ch = tid >> 4;
    if (ch >= B_ * DI_) return;
    int d = ch % DI_;
    int b = ch / DI_;
    float An   = -__expf(A_log[d * DS_ + n]);
    float bias = dt_bias[d];
    float Dd   = Dv[d];
    const float*  dtrow = dt  + ((long)b * DI_ + d) * L_;
    const __bf16* urow  = xcT + (long)b * L_ * DI_ + d;
    const float*  bcrow = bcT + (long)b * L_ * 32;
    const float*  zrow  = xz  + ((long)b * 2 * DI_ + DI_ + d) * L_;
    float* yrow = y + ((long)b * DI_ + d) * L_;
    float h = 0.f;
    for (int t = 0; t < L_; ++t) {
        float dtv   = dtrow[t] + bias;
        float delta = (dtv > 20.f) ? dtv : __logf(1.f + __expf(dtv));  // softplus
        float u  = (float)urow[(long)t * DI_];
        float Bt = bcrow[t * 32 + n];
        float Ct = bcrow[t * 32 + 16 + n];
        h = __expf(delta * An) * h + (delta * u) * Bt;
        float p = h * Ct;
        p += __shfl_xor(p, 1);
        p += __shfl_xor(p, 2);
        p += __shfl_xor(p, 4);
        p += __shfl_xor(p, 8);
        if (n == 0) {
            int lz = reverse ? (L_ - 1 - t) : t;
            float yv = (p + u * Dd) * siluf(zrow[lz]);
            if (reverse) yrow[lz] += yv;   // add onto branch-a result
            else         yrow[t]  = yv;
        }
    }
}

// ---------------------------------------------------------------------------
// y [B,DI,L] f32 -> yT [B,L,DI] bf16 (K-contiguous A operand for out_proj GEMM)
// ---------------------------------------------------------------------------
__global__ void y_transpose_kernel(const float* __restrict__ y, __bf16* __restrict__ yT) {
    long idx = (long)blockIdx.x * blockDim.x + threadIdx.x;
    long tot = (long)B_ * L_ * DI_;
    if (idx >= tot) return;
    int d = (int)(idx % DI_);
    int l = (int)((idx / DI_) % L_);
    int b = (int)(idx / ((long)DI_ * L_));
    yT[idx] = (__bf16)y[((long)b * DI_ + d) * L_ + l];
}

// ---------------------------------------------------------------------------
extern "C" void kernel_launch(void* const* d_in, const int* in_sizes, int n_in,
                              void* d_out, int out_size, void* d_ws, size_t ws_size,
                              hipStream_t stream) {
    const float* hidden      = (const float*)d_in[0];
    const float* in_proj_w   = (const float*)d_in[1];
    const float* out_proj_w  = (const float*)d_in[2];
    const float* conv_w[2]   = { (const float*)d_in[3], (const float*)d_in[5] };
    const float* conv_b[2]   = { (const float*)d_in[4], (const float*)d_in[6] };
    const float* x_proj_w[2] = { (const float*)d_in[7], (const float*)d_in[8] };
    const float* dt_proj_w[2]= { (const float*)d_in[9], (const float*)d_in[10] };
    const float* dt_bias[2]  = { (const float*)d_in[11], (const float*)d_in[12] };
    const float* A_log[2]    = { (const float*)d_in[13], (const float*)d_in[14] };
    const float* Dvec[2]     = { (const float*)d_in[15], (const float*)d_in[16] };

    char*  ws  = (char*)d_ws;
    size_t off = 0;
    auto alloc = [&](size_t bytes) -> char* {
        char* p = ws + off;
        off = (off + bytes + 255) & ~(size_t)255;
        return p;
    };

    __bf16* hbf   = (__bf16*)alloc((size_t)B_ * L_ * DM_ * 2);
    __bf16* w_in  = (__bf16*)alloc((size_t)2 * DI_ * DM_ * 2);
    __bf16* w_out = (__bf16*)alloc((size_t)DM_ * DI_ * 2);
    __bf16* w_xp[2] = { (__bf16*)alloc((size_t)96 * DI_ * 2),
                        (__bf16*)alloc((size_t)96 * DI_ * 2) };
    __bf16* w_dt[2] = { (__bf16*)alloc((size_t)DI_ * 64 * 2),
                        (__bf16*)alloc((size_t)DI_ * 64 * 2) };
    float*  xz   = (float*)alloc((size_t)B_ * 2 * DI_ * L_ * 4);
    __bf16* xcT  = (__bf16*)alloc((size_t)B_ * L_ * DI_ * 2);   // reused per branch
    float*  dbl  = (float*)alloc((size_t)B_ * 96 * L_ * 4);     // reused per branch
    __bf16* dtlT = (__bf16*)alloc((size_t)B_ * L_ * 64 * 2);    // reused per branch
    float*  bcT  = (float*)alloc((size_t)B_ * L_ * 32 * 4);     // reused per branch
    float*  dtf  = (float*)alloc((size_t)B_ * DI_ * L_ * 4);    // reused per branch
    float*  yacc = (float*)alloc((size_t)B_ * DI_ * L_ * 4);
    __bf16* yT   = (__bf16*)alloc((size_t)B_ * L_ * DI_ * 2);

    auto nblk = [](long n, int bs) { return dim3((unsigned)((n + bs - 1) / bs)); };

    auto gemm = [&](const __bf16* A, const __bf16* Bmat, float* C,
                    int M, int N, int K, long sAb, long sBb) {
        int  tm = M / 32, tn = N / 32;
        long total = (long)B_ * tm * tn;
        long nb = (total + 7) / 8;  // 8 waves (one 32x32 tile each) per 256-thread block
        gemm_nt_bf16<<<dim3((unsigned)nb), dim3(256), 0, stream>>>(
            A, Bmat, C, M, N, K, sAb, sBb, tm, tn, total);
    };

    // --- precision conversions / padding ---
    pad2d_f32_bf16<<<nblk((long)B_ * L_ * DM_, 256), 256, 0, stream>>>(
        hidden, hbf, B_ * L_, DM_, B_ * L_, DM_);
    pad2d_f32_bf16<<<nblk((long)2 * DI_ * DM_, 256), 256, 0, stream>>>(
        in_proj_w, w_in, 2 * DI_, DM_, 2 * DI_, DM_);
    pad2d_f32_bf16<<<nblk((long)DM_ * DI_, 256), 256, 0, stream>>>(
        out_proj_w, w_out, DM_, DI_, DM_, DI_);
    for (int br = 0; br < 2; ++br) {
        pad2d_f32_bf16<<<nblk((long)96 * DI_, 256), 256, 0, stream>>>(
            x_proj_w[br], w_xp[br], DTR_ + 2 * DS_, DI_, 96, DI_);
        pad2d_f32_bf16<<<nblk((long)DI_ * 64, 256), 256, 0, stream>>>(
            dt_proj_w[br], w_dt[br], DI_, DTR_, DI_, 64);
    }

    // --- in_proj: xz[b, 3072, L] = in_proj_w x hidden^T ---
    gemm(w_in, hbf, xz, 2 * DI_, L_, DM_, 0, (long)L_ * DM_);

    // --- two branches (branch buffers reused; stream-ordered) ---
    for (int br = 0; br < 2; ++br) {
        conv_silu_kernel<<<nblk((long)B_ * DI_ * L_, 256), 256, 0, stream>>>(
            xz, conv_w[br], conv_b[br], xcT, br);
        // x_proj (M padded 80->96): dbl[b,96,L]
        gemm(w_xp[br], xcT, dbl, 96, L_, DI_, 0, (long)L_ * DI_);
        pack_dbl_kernel<<<nblk((long)B_ * L_ * 64, 256), 256, 0, stream>>>(dbl, dtlT, bcT);
        // dt_proj (K padded 48->64): dt[b,1536,L]
        gemm(w_dt[br], dtlT, dtf, DI_, L_, 64, 0, (long)L_ * 64);
        scan_kernel<<<nblk((long)B_ * DI_ * DS_, 256), 256, 0, stream>>>(
            dtf, xcT, bcT, xz, A_log[br], dt_bias[br], Dvec[br], yacc, br);
    }

    // --- out_proj: out[b, l, o] = yT x out_proj_w^T ---
    y_transpose_kernel<<<nblk((long)B_ * L_ * DI_, 256), 256, 0, stream>>>(yacc, yT);
    gemm(yT, w_out, (float*)d_out, L_, DM_, DI_, (long)L_ * DI_, 0);

    (void)in_sizes; (void)n_in; (void)out_size; (void)ws_size;
}
